// LocalHOPBlock_55688545960150
// MI455X (gfx1250) — matrix operand order repaired and checked by reading, not compiled, not run on hardware
//
#include <hip/hip_runtime.h>
#include <hip/hip_bf16.h>

// ---------------- problem constants ----------------
static constexpr int Hh   = 512;
static constexpr int Ww   = 512;
static constexpr int HWp  = Hh * Ww;        // 262144
static constexpr int NB   = 2;              // batch
static constexpr int CG   = 128;            // guidance in-channels
static constexpr int CGH  = 64;             // guidance out-channels
static constexpr int PADR = 3;
static constexpr int HALO = 22;             // 16 + 2*PADR
static constexpr int HALO2 = HALO * HALO;   // 484

typedef __attribute__((ext_vector_type(16))) __bf16         v16bf;
typedef __attribute__((ext_vector_type(16))) unsigned short v16u;
typedef __attribute__((ext_vector_type(8)))  float          v8f;

union bfvec  { v16u u; v16bf b; };
union bfvecq { uint4 q[2]; v16u u; v16bf b; };   // for 2x b128 loads

__device__ __forceinline__ unsigned short f2bf_u(float x) {
    union { __bf16 h; unsigned short s; } c;
    c.h = (__bf16)x;                       // native v_cvt f32->bf16
    return c.s;
}
__device__ __forceinline__ float bfu2f(unsigned short b) {
    return __uint_as_float(((unsigned)b) << 16);
}
__device__ __forceinline__ int refl(int p, int n) {
    p = (p < 0) ? -p : p;
    return (p >= n) ? (2 * n - 2 - p) : p;
}
__device__ __forceinline__ v8f wmma_bf16(v16bf a, v16bf b, v8f c) {
    // D = A(16x32) * B(32x16) + C, fp32 accumulate
    return __builtin_amdgcn_wmma_f32_16x16x32_bf16(false, a, false, b, (short)0, c, false, false);
}

// ================= K0: sigma + fragment prep + zero stats =================
__global__ void k0_prep(const float* __restrict__ wg, const float* __restrict__ w_bar,
                        const float* __restrict__ u,
                        unsigned short* __restrict__ wsnA,
                        unsigned short* __restrict__ wgA,
                        float* __restrict__ stats)
{
    __shared__ float sv[32];
    __shared__ float su[32];
    __shared__ float red[2];
    const int t = threadIdx.x;

    if (t < 32) {                       // v1_raw = Wb^T u
        float a = 0.f;
        for (int o = 0; o < 32; ++o) a += w_bar[o * 32 + t] * u[o];
        sv[t] = a;
    }
    __syncthreads();
    if (t == 0) {
        float n = 0.f;
        for (int c = 0; c < 32; ++c) n += sv[c] * sv[c];
        red[0] = sqrtf(n) + 1e-12f;
    }
    __syncthreads();
    if (t < 32) sv[t] = sv[t] / red[0];  // v1
    __syncthreads();
    if (t < 32) {                        // su = Wb v1
        float a = 0.f;
        for (int c = 0; c < 32; ++c) a += w_bar[t * 32 + c] * sv[c];
        su[t] = a;
    }
    __syncthreads();
    if (t == 0) {                        // sigma = u1 . (Wb v1) = |su|^2 / (|su|+eps)
        float n2 = 0.f;
        for (int o = 0; o < 32; ++o) n2 += su[o] * su[o];
        red[1] = n2 / (sqrtf(n2) + 1e-12f);
    }
    __syncthreads();
    const float isig = 1.0f / red[1];

    // w_sn A-fragments (2 M-tiles, K=32) in contiguous-K order (B operands use the same):
    // lane 0-15: M=lane, K = kb+i ; kb = (lane>>4)*16
    for (int idx = t; idx < 2 * 32 * 16; idx += blockDim.x) {
        int i = idx & 15;
        int lane = (idx >> 4) & 31;
        int mt = idx >> 9;
        int m = mt * 16 + (lane & 15);
        int kb = (lane >> 4) * 16;
        int k = kb + i;
        wsnA[idx] = f2bf_u(w_bar[m * 32 + k] * isig);
    }
    // wg A-fragments (4 M-tiles x 4 K-steps)
    for (int idx = t; idx < 4 * 4 * 32 * 16; idx += blockDim.x) {
        int i = idx & 15;
        int lane = (idx >> 4) & 31;
        int ks = (idx >> 9) & 3;
        int mt = idx >> 11;
        int m = mt * 16 + (lane & 15);
        int kb = (lane >> 4) * 16;
        int kk = kb + i;
        wgA[idx] = f2bf_u(wg[m * CG + ks * 32 + kk]);
    }
    if (t < 64) stats[t] = 0.f;          // sums[32] | sumsq[32]
}

// ================= K1: guidance conv (WMMA) + L2 normalize ================
__global__ __launch_bounds__(256)
void k1_guidance(const float* __restrict__ f, const float* __restrict__ bg,
                 const unsigned short* __restrict__ wgA, unsigned* __restrict__ gn_pk)
{
    const int lane = threadIdx.x & 31;
    const int wave = threadIdx.x >> 5;
    const int h = blockIdx.y;
    const int b = blockIdx.z;
    const int p = blockIdx.x * 128 + wave * 16 + (lane & 15);
    const int hi = lane >> 4;
    const v16bf* wgAv = (const v16bf*)wgA;

    v8f acc0 = {}, acc1 = {}, acc2 = {}, acc3 = {};
#pragma unroll
    for (int ks = 0; ks < 4; ++ks) {
        bfvec bb;
        const int kbase = ks * 32 + hi * 16;       // dense-B: lanes 0-15 K=0-15, 16-31 K=16-31
#pragma unroll
        for (int i = 0; i < 16; ++i)
            bb.b[i] = (__bf16)f[((b * CG + kbase + i) * Hh + h) * Ww + p];
        v16bf a0 = wgAv[(0 * 4 + ks) * 32 + lane];
        v16bf a1 = wgAv[(1 * 4 + ks) * 32 + lane];
        v16bf a2 = wgAv[(2 * 4 + ks) * 32 + lane];
        v16bf a3 = wgAv[(3 * 4 + ks) * 32 + lane];
        acc0 = wmma_bf16(a0, bb.b, acc0);
        acc1 = wmma_bf16(a1, bb.b, acc1);
        acc2 = wmma_bf16(a2, bb.b, acc2);
        acc3 = wmma_bf16(a3, bb.b, acc3);
    }
    // bias + channel-wise L2 norm over 64 channels (32 here + 32 in lane^16)
    const int cb = hi * 8;                 // C-tile: reg r -> channel mt*16 + cb + r
    float vals[32];
    float ss = 0.f;
#pragma unroll
    for (int r = 0; r < 8; ++r) {
        float v0 = acc0[r] + bg[ 0 + cb + r];
        float v1 = acc1[r] + bg[16 + cb + r];
        float v2 = acc2[r] + bg[32 + cb + r];
        float v3 = acc3[r] + bg[48 + cb + r];
        vals[r] = v0; vals[8 + r] = v1; vals[16 + r] = v2; vals[24 + r] = v3;
        ss += v0 * v0 + v1 * v1 + v2 * v2 + v3 * v3;
    }
    ss += __shfl_xor(ss, 16, 32);          // partner lane holds the other 32 channels
    const float sc = 1.0f / fmaxf(sqrtf(ss), 1e-4f);
#pragma unroll
    for (int mt = 0; mt < 4; ++mt) {
#pragma unroll
        for (int r = 0; r < 8; r += 2) {
            int m = mt * 16 + cb + r;      // even channel; pair (m, m+1)
            unsigned lo = f2bf_u(vals[mt * 8 + r] * sc);
            unsigned hv = f2bf_u(vals[mt * 8 + r + 1] * sc);
            gn_pk[((b * (CGH / 2) + (m >> 1)) * Hh + h) * Ww + p] = lo | (hv << 16);
        }
    }
}

// ====== K2: 49-tap correlation + softmax + alpha aggregation (LDS tile) ======
// y written pixel-major: yt[(b*HW + pix)*32 + c]  (bf16) so K3/K4 can b128-load
__global__ __launch_bounds__(256)
void k2_correlate(const unsigned* __restrict__ gn_pk, const float* __restrict__ alpha,
                  unsigned short* __restrict__ yt)
{
    extern __shared__ char smem[];
    unsigned* gnS = (unsigned*)smem;                                // 32 pairs x 22x22
    float*    alS = (float*)(smem + 32 * HALO2 * sizeof(unsigned)); // 32 ch x 22x22
    const int tid = threadIdx.x;
    const int tx = tid & 15, ty = tid >> 4;
    const int x0 = blockIdx.x * 16, y0 = blockIdx.y * 16;
    const int b = blockIdx.z;

    for (int idx = tid; idx < 32 * HALO2; idx += 256) {
        int cp = idx / HALO2, rem = idx % HALO2;
        int yy = rem / HALO, xx = rem % HALO;
        int gy = refl(y0 + yy - PADR, Hh);
        int gx = refl(x0 + xx - PADR, Ww);
        gnS[idx] = gn_pk[((b * 32 + cp) * Hh + gy) * Ww + gx];
        alS[idx] = alpha[((b * 32 + cp) * Hh + gy) * Ww + gx];
    }
    __syncthreads();

    float s[49];
#pragma unroll
    for (int k = 0; k < 49; ++k) s[k] = 0.f;
    for (int cp = 0; cp < 32; ++cp) {                 // rolled: keeps s[] reg-resident
        const unsigned* g = gnS + cp * HALO2;
        unsigned ce = g[(ty + PADR) * HALO + (tx + PADR)];
        float cx = bfu2f((unsigned short)(ce & 0xffffu));
        float cy = bfu2f((unsigned short)(ce >> 16));
#pragma unroll
        for (int k = 0; k < 49; ++k) {
            if (k == 24) continue;                    // center handled below
            unsigned nb = g[(ty + k / 7) * HALO + (tx + k % 7)];
            s[k] += cx * bfu2f((unsigned short)(nb & 0xffffu))
                  + cy * bfu2f((unsigned short)(nb >> 16));
        }
    }
    s[24] = -10000.f;
    float mx = s[0];
#pragma unroll
    for (int k = 1; k < 49; ++k) mx = fmaxf(mx, s[k]);
    float sum = 0.f;
#pragma unroll
    for (int k = 0; k < 49; ++k) { s[k] = __expf(s[k] - mx); sum += s[k]; }
    const float inv = 1.0f / sum;
#pragma unroll
    for (int k = 0; k < 49; ++k) s[k] *= inv;

    const int pixbase = ((b * HWp) + (y0 + ty) * Ww + (x0 + tx)) * 32;
    for (int c = 0; c < 32; c += 2) {                 // pair channels -> b32 stores
        const float* a0 = alS + c * HALO2;
        const float* a1 = a0 + HALO2;
        float acc0 = 0.f, acc1 = 0.f;
#pragma unroll
        for (int k = 0; k < 49; ++k) {
            if (k == 24) continue;                    // skip center tap
            int o = (ty + k / 7) * HALO + (tx + k % 7);
            acc0 += s[k] * a0[o];
            acc1 += s[k] * a1[o];
        }
        unsigned pk = (unsigned)f2bf_u(acc0) | ((unsigned)f2bf_u(acc1) << 16);
        *((unsigned*)(yt + pixbase + c)) = pk;
    }
}

// ========= K3: SN-conv (WMMA) per-channel sum / sumsq for BatchNorm =========
__global__ __launch_bounds__(256)
void k3_stats(const unsigned short* __restrict__ yt,
              const unsigned short* __restrict__ wsnA,
              float* __restrict__ stats)
{
    const int lane = threadIdx.x & 31;
    const int gw = (int)((blockIdx.x * blockDim.x + threadIdx.x) >> 5);
    const int nw = (int)((gridDim.x * blockDim.x) >> 5);
    const v16bf* wf = (const v16bf*)wsnA;
    v16bf a0 = wf[lane], a1 = wf[32 + lane];
    const int kb = (lane >> 4) * 16;
    float sum0[8] = {}, sq0[8] = {}, sum1[8] = {}, sq1[8] = {};
    const int nChunks = (NB * HWp) / 16;

    for (int ch = gw; ch < nChunks; ch += nw) {
        int pix = ch * 16 + (lane & 15);
        const uint4* p4 = (const uint4*)(yt + pix * 32 + kb);  // 32B contiguous
        bfvecq bb;
        bb.q[0] = p4[0];
        bb.q[1] = p4[1];
        v8f c0 = {}, c1 = {};
        c0 = wmma_bf16(a0, bb.b, c0);
        c1 = wmma_bf16(a1, bb.b, c1);
#pragma unroll
        for (int r = 0; r < 8; ++r) {
            float v = c0[r]; sum0[r] += v; sq0[r] += v * v;
            v = c1[r];       sum1[r] += v; sq1[r] += v * v;
        }
    }
    // lanes 0-15 all hold channel (mt*16 + r); lanes 16-31 hold (mt*16 + 8 + r)
#pragma unroll
    for (int r = 0; r < 8; ++r) {
#pragma unroll
        for (int m = 1; m < 16; m <<= 1) {
            sum0[r] += __shfl_xor(sum0[r], m, 32);
            sq0[r]  += __shfl_xor(sq0[r],  m, 32);
            sum1[r] += __shfl_xor(sum1[r], m, 32);
            sq1[r]  += __shfl_xor(sq1[r],  m, 32);
        }
    }
    if ((lane & 15) == 0) {
        const int off = (lane >> 4) * 8;
#pragma unroll
        for (int r = 0; r < 8; ++r) {
            atomicAdd(stats + (off + r),           sum0[r]);
            atomicAdd(stats + 32 + (off + r),      sq0[r]);
            atomicAdd(stats + (16 + off + r),      sum1[r]);
            atomicAdd(stats + 32 + (16 + off + r), sq1[r]);
        }
    }
}

// ====== K4: SN-conv (WMMA) + BatchNorm finalize + residual -> output ======
__global__ __launch_bounds__(256)
void k4_output(const unsigned short* __restrict__ yt,
               const unsigned short* __restrict__ wsnA,
               const float* __restrict__ stats,
               const float* __restrict__ alpha,
               float* __restrict__ out)
{
    const int lane = threadIdx.x & 31;
    const int gw = (int)((blockIdx.x * blockDim.x + threadIdx.x) >> 5);
    const int nw = (int)((gridDim.x * blockDim.x) >> 5);
    const v16bf* wf = (const v16bf*)wsnA;
    v16bf a0 = wf[lane], a1 = wf[32 + lane];
    const int kb = (lane >> 4) * 16;
    const int off = (lane >> 4) * 8;
    const float invN = 1.0f / (float)(NB * HWp);

    float mean0[8], sc0[8], mean1[8], sc1[8];
#pragma unroll
    for (int r = 0; r < 8; ++r) {
        int m = off + r;
        float mu = stats[m] * invN;
        float var = stats[32 + m] * invN - mu * mu;
        mean0[r] = mu; sc0[r] = 0.1f * rsqrtf(var + 1e-5f);
        m = 16 + off + r;
        mu = stats[m] * invN;
        var = stats[32 + m] * invN - mu * mu;
        mean1[r] = mu; sc1[r] = 0.1f * rsqrtf(var + 1e-5f);
    }

    const int nChunks = (NB * HWp) / 16;
    for (int ch = gw; ch < nChunks; ch += nw) {
        int pix = ch * 16 + (lane & 15);
        int bi = pix >> 18;                    // / HWp
        int pp = pix & (HWp - 1);
        const uint4* p4 = (const uint4*)(yt + pix * 32 + kb);
        bfvecq bb;
        bb.q[0] = p4[0];
        bb.q[1] = p4[1];
        v8f c0 = {}, c1 = {};
        c0 = wmma_bf16(a0, bb.b, c0);
        c1 = wmma_bf16(a1, bb.b, c1);
#pragma unroll
        for (int r = 0; r < 8; ++r) {
            int m = off + r;
            int idx = (bi * 32 + m) * HWp + pp;
            out[idx] = (c0[r] - mean0[r]) * sc0[r] + alpha[idx];
            m = 16 + off + r;
            idx = (bi * 32 + m) * HWp + pp;
            out[idx] = (c1[r] - mean1[r]) * sc1[r] + alpha[idx];
        }
    }
}

// =========================== host entry ===========================
extern "C" void kernel_launch(void* const* d_in, const int* in_sizes, int n_in,
                              void* d_out, int out_size, void* d_ws, size_t ws_size,
                              hipStream_t stream)
{
    (void)in_sizes; (void)n_in; (void)out_size; (void)ws_size;
    const float* f     = (const float*)d_in[0];
    const float* alpha = (const float*)d_in[1];
    const float* wg    = (const float*)d_in[2];
    const float* bg    = (const float*)d_in[3];
    const float* w_bar = (const float*)d_in[4];
    const float* u     = (const float*)d_in[5];
    // d_in[6] = v : unused by the reference computation
    float* out = (float*)d_out;

    char* ws = (char*)d_ws;
    unsigned*       gn_pk = (unsigned*)ws;                                  // 64 MiB
    unsigned short* yt    = (unsigned short*)(ws + ((size_t)64 << 20));     // 32 MiB, pixel-major
    float*          stats = (float*)(ws + ((size_t)96 << 20));              // 256 B
    unsigned short* wsnA  = (unsigned short*)(ws + ((size_t)96 << 20) + 1024);
    unsigned short* wgA   = (unsigned short*)(ws + ((size_t)96 << 20) + 4096);

    k0_prep<<<1, 256, 0, stream>>>(wg, w_bar, u, wsnA, wgA, stats);
    k1_guidance<<<dim3(Ww / 128, Hh, NB), 256, 0, stream>>>(f, bg, wgA, gn_pk);
    const size_t smem = (size_t)32 * HALO2 * sizeof(unsigned)
                      + (size_t)32 * HALO2 * sizeof(float);                 // ~124 KB (<320 KB WGP LDS)
    k2_correlate<<<dim3(Ww / 16, Hh / 16, NB), 256, smem, stream>>>(gn_pk, alpha, yt);
    k3_stats<<<dim3(128), 256, 0, stream>>>(yt, wsnA, stats);
    k4_output<<<dim3(256), 256, 0, stream>>>(yt, wsnA, stats, alpha, out);
}
